// LSTM_65257733096081
// MI455X (gfx1250) — compile-verified, matrix operand order checked by reading, not run on hardware
//
#include <hip/hip_runtime.h>
#include <cstdint>
#include <cstddef>

// ---------------------------------------------------------------------------
// LSTM (B=64, T=512, I=H=1024) for gfx1250 (MI455X), bf16 WMMA + async→LDS.
//
// Pipeline (all on `stream`):
//   P1 pack_w : pack [Wi;Wh] (2048x4096 f32) -> bf16 in per-lane WMMA
//               B-fragment order, grouped per owning workgroup.
//   P2 bias_k : bias[g*1024+j] = b_i[g][j] + b_h[g][j]
//   P3 xcvt   : x f32 -> bf16
//   P4 init_k : zero h ping buffer, zero barrier counter
//   P5 xproj  : x_proj = x @ Wi + b  (bf16 WMMA, f32 acc), 128-row tiles
//               (2 timesteps / WG => each B fragment feeds 2 WMMAs),
//               stored bf16 in packed C-fragment order.
//   P6 scan   : persistent kernel, 64 WGs x 128 thr; Wh slice (128KB) LDS-
//               resident for all 512 steps; c in registers; h ping-pong in
//               global; atomic-counter device barrier per step.
// ---------------------------------------------------------------------------

typedef __attribute__((ext_vector_type(16))) __bf16 v16bf;
typedef __attribute__((ext_vector_type(8)))  __bf16 v8bf;
typedef __attribute__((ext_vector_type(4)))  __bf16 v4bf;
typedef __attribute__((ext_vector_type(8)))  float  v8f;

union V16 { v16bf v; v8bf h[2]; };

#define NB   64      // batch
#define NT   512     // time steps
#define NI   1024    // input dim
#define NH   1024    // hidden dim
#define NWG  64      // workgroups in scan kernel (16 H-cols each)

// ---- workspace layout (bytes) ----
static constexpr size_t W_ELEMS  = 64ull * 64 * 4 * 512;        // 8,388,608 bf16 (packed [Wi;Wh])
static constexpr size_t OFF_W    = 0;
static constexpr size_t OFF_BIAS = OFF_W + W_ELEMS * 2;         // 4096 f32
static constexpr size_t OFF_XB   = OFF_BIAS + 4096 * 4;         // x bf16
static constexpr size_t XB_ELEMS = 64ull * 512 * 1024;
static constexpr size_t OFF_XP   = OFF_XB + XB_ELEMS * 2;       // packed x_proj bf16
static constexpr size_t XP_ELEMS = 512ull * 64 * 4096;
static constexpr size_t OFF_H0   = OFF_XP + XP_ELEMS * 2;       // h ping (bf16 64x1024)
static constexpr size_t OFF_H1   = OFF_H0 + 65536ull * 2;       // h pong
static constexpr size_t OFF_CNT  = OFF_H1 + 65536ull * 2;       // u32 barrier counter

// ---------------------------------------------------------------------------
// device helpers
// ---------------------------------------------------------------------------
__device__ __forceinline__ v8f wmma_bf16(v16bf a, v16bf b, v8f c) {
  return __builtin_amdgcn_wmma_f32_16x16x32_bf16(false, a, false, b,
                                                 (short)0, c, false, false);
}

// one lane moves 16B global -> LDS asynchronously (ASYNCcnt)
__device__ __forceinline__ void async_g2l_b128(uint32_t lds_byte, const void* g) {
  asm volatile("global_load_async_to_lds_b128 %0, %1, off"
               :: "v"(lds_byte), "v"((unsigned long long)(uintptr_t)g)
               : "memory");
}
__device__ __forceinline__ void wait_async_le8() {
  asm volatile("s_wait_asynccnt 0x8" ::: "memory");
}
__device__ __forceinline__ void wait_async_le16() {
  asm volatile("s_wait_asynccnt 0x10" ::: "memory");
}
__device__ __forceinline__ void wait_async_0() {
  asm volatile("s_wait_asynccnt 0x0" ::: "memory");
}

// stage a 64-row x 128-col bf16 tile (16KB): 128 threads x 8 async b128 ops
__device__ __forceinline__ void stage_tile(uint32_t lds_base, const char* src0,
                                           size_t row_stride_b, int tid) {
#pragma unroll
  for (int i = 0; i < 8; ++i) {
    int s   = i * 128 + tid;       // 0..1023
    int row = s >> 4;              // 0..63
    int seg = s & 15;              // 16B segment within 256B row
    async_g2l_b128(lds_base + (uint32_t)(row * 256 + seg * 16),
                   src0 + (size_t)row * row_stride_b + (size_t)seg * 16);
  }
}

__device__ __forceinline__ float sigmoid_f(float x) {
  return 1.0f / (1.0f + __expf(-x));
}
__device__ __forceinline__ float tanh_f(float x) {
  float e = __expf(-2.0f * fabsf(x));      // in (0,1], overflow-safe
  float r = (1.0f - e) / (1.0f + e);
  return copysignf(r, x);
}

// k-offset inside a 32-deep WMMA tile for fragment element j, lane-half h
__device__ __forceinline__ int koff_map(int j, int h) {
  return (j < 8) ? (h * 8 + j) : (16 + h * 8 + (j - 8));
}

// ---------------------------------------------------------------------------
// P1: pack [Wi;Wh] -> bf16 per-lane B-fragment order.
// elem o = wg*131072 + (kt*4+ct)*512 + l*16 + j  maps to  W[k][col]
//   k = kt*32 + koff(j, l>>4), col = wg*16 + (l&15), gate = ct
// kt 0..31 = Wi (k 0..1023), kt 32..63 = Wh (k 1024..2047)
// ---------------------------------------------------------------------------
__global__ void pack_w_k(const float* __restrict__ wii, const float* __restrict__ wif,
                         const float* __restrict__ wig, const float* __restrict__ wio,
                         const float* __restrict__ whi, const float* __restrict__ whf,
                         const float* __restrict__ whg, const float* __restrict__ who,
                         unsigned short* __restrict__ wws) {
  unsigned o = blockIdx.x * 256u + threadIdx.x;   // < 8,388,608
  unsigned wg = o >> 17;
  unsigned r  = o & 131071u;
  unsigned frag = r >> 9, kt = frag >> 2, ct = frag & 3u;
  unsigned e = r & 511u, l = e >> 4, j = e & 15u;
  int k   = (int)(kt * 32u) + koff_map((int)j, (int)(l >> 4));
  int col = (int)(wg * 16u + (l & 15u));
  const float* wi[4] = {wii, wif, wig, wio};
  const float* wh[4] = {whi, whf, whg, who};
  float v = (k < 1024) ? wi[ct][(size_t)k * 1024 + col]
                       : wh[ct][(size_t)(k - 1024) * 1024 + col];
  __bf16 b = (__bf16)v;
  wws[o] = *(unsigned short*)&b;
}

__global__ void bias_k(const float* __restrict__ bii, const float* __restrict__ bif,
                       const float* __restrict__ big, const float* __restrict__ bio,
                       const float* __restrict__ bhi, const float* __restrict__ bhf,
                       const float* __restrict__ bhg, const float* __restrict__ bho,
                       float* __restrict__ bias) {
  int idx = blockIdx.x * 256 + threadIdx.x;
  if (idx >= 4096) return;
  int g = idx >> 10, j = idx & 1023;
  const float* bi[4] = {bii, bif, big, bio};
  const float* bh[4] = {bhi, bhf, bhg, bho};
  bias[idx] = bi[g][j] + bh[g][j];
}

__global__ void xcvt_k(const float* __restrict__ x, unsigned short* __restrict__ xb) {
  size_t i4 = ((size_t)blockIdx.x * 256 + threadIdx.x) * 4;   // total 33,554,432 elems
  float4 v = *(const float4*)&x[i4];
  v4bf o = {(__bf16)v.x, (__bf16)v.y, (__bf16)v.z, (__bf16)v.w};
  *(v4bf*)&xb[i4] = o;
}

__global__ void init_k(unsigned short* __restrict__ h0, unsigned* __restrict__ cnt) {
  int idx = blockIdx.x * 256 + threadIdx.x;
  if (idx < 65536) h0[idx] = 0;       // bf16 +0.0
  if (idx == 0)    *cnt   = 0;
}

// ---------------------------------------------------------------------------
// P5: x_proj = x @ Wi + bias  -> bf16 packed C-fragments
// grid: 16384 blocks (tp = bid>>6 covers t0=2*tp and t0+1, wg = bid&63),
// 128 threads (4 waves). Two 64-row A tiles share every B fragment.
// ---------------------------------------------------------------------------
__global__ void __launch_bounds__(128)
xproj_k(const unsigned short* __restrict__ xb_u, const unsigned short* __restrict__ wws_u,
        const float* __restrict__ bias, unsigned short* __restrict__ xp_u) {
  const __bf16* xb  = (const __bf16*)xb_u;
  const __bf16* wws = (const __bf16*)wws_u;
  __bf16* xp        = (__bf16*)xp_u;

  const int bid = blockIdx.x;
  const int tp  = bid >> 6;            // timestep pair
  const int t0  = tp * 2;
  const int wg  = bid & 63;
  const int tid = threadIdx.x;
  const int w   = tid >> 5;            // wave = row block (16 rows)
  const int l   = tid & 31;
  const int l16 = l & 15;
  const int h8  = (l >> 4) * 8;

  __shared__ __bf16 As[2][2][64 * 128];          // [buf][t-tile] : 64KB
  uint32_t asb[2][2] = {
    {(uint32_t)(uintptr_t)&As[0][0][0], (uint32_t)(uintptr_t)&As[0][1][0]},
    {(uint32_t)(uintptr_t)&As[1][0][0], (uint32_t)(uintptr_t)&As[1][1][0]}};

  v8f acc0[4], acc1[4];
#pragma unroll
  for (int ct = 0; ct < 4; ++ct) {
    float bs = bias[ct * 1024 + wg * 16 + l16];
#pragma unroll
    for (int r = 0; r < 8; ++r) { acc0[ct][r] = bs; acc1[ct][r] = bs; }
  }

  const __bf16* wbase = wws + (size_t)wg * 131072;   // this WG's packed Wi/Wh
  // row m of tile tt lives at xb + m*512*1024 + (t0+tt)*1024 (elems)
  const char* xrow0 = (const char*)xb + (size_t)t0 * 1024 * 2;
  const size_t XSTRIDE = 1048576;                    // 512*1024*2 bytes

  stage_tile(asb[0][0], xrow0,        XSTRIDE, tid);
  stage_tile(asb[0][1], xrow0 + 2048, XSTRIDE, tid);
  for (int c = 0; c < 8; ++c) {                      // K = 1024 in 128-chunks
    if (c < 7) {
      const char* nsrc = xrow0 + (size_t)(c + 1) * 256;
      stage_tile(asb[(c + 1) & 1][0], nsrc,        XSTRIDE, tid);
      stage_tile(asb[(c + 1) & 1][1], nsrc + 2048, XSTRIDE, tid);
      wait_async_le16();
    } else {
      wait_async_0();
    }
    __syncthreads();
    const __bf16* A0 = &As[c & 1][0][0];
    const __bf16* A1 = &As[c & 1][1][0];
#pragma unroll
    for (int kt = 0; kt < 4; ++kt) {
      int row = w * 16 + l16;
      V16 a0, a1;
      a0.h[0] = *(const v8bf*)&A0[row * 128 + kt * 32 + h8];
      a0.h[1] = *(const v8bf*)&A0[row * 128 + kt * 32 + h8 + 16];
      a1.h[0] = *(const v8bf*)&A1[row * 128 + kt * 32 + h8];
      a1.h[1] = *(const v8bf*)&A1[row * 128 + kt * 32 + h8 + 16];
      int ktg = c * 4 + kt;                          // Wi tiles: kt 0..31
#pragma unroll
      for (int ct = 0; ct < 4; ++ct) {
        size_t bo = ((size_t)(ktg * 4 + ct)) * 512 + (size_t)l * 16;
        V16 b;
        b.h[0] = *(const v8bf*)&wbase[bo];
        b.h[1] = *(const v8bf*)&wbase[bo + 8];
        acc0[ct] = wmma_bf16(a0.v, b.v, acc0[ct]);   // B reused from registers
        acc1[ct] = wmma_bf16(a1.v, b.v, acc1[ct]);
      }
    }
    __syncthreads();
  }

  // store packed bf16 C-fragments: [t][wg][w][ct][lane][8]
#pragma unroll
  for (int ct = 0; ct < 4; ++ct) {
    v8bf o0, o1;
#pragma unroll
    for (int r = 0; r < 8; ++r) { o0[r] = (__bf16)acc0[ct][r]; o1[r] = (__bf16)acc1[ct][r]; }
    size_t base = ((size_t)wg * 4 + w) * 4 + ct;
    size_t idx0 = ((size_t)t0 * 1024 + base) * 256 + (size_t)l * 8;       // t0*64*16*... == ((t*64+wg)*4+w)*4+ct)*256
    size_t idx1 = ((size_t)(t0 + 1) * 1024 + base) * 256 + (size_t)l * 8;
    *(v8bf*)&xp[idx0] = o0;
    *(v8bf*)&xp[idx1] = o1;
  }
}

// ---------------------------------------------------------------------------
// P6: persistent recurrent scan. 64 WGs x 128 threads; Wh slice in LDS.
// ---------------------------------------------------------------------------
__global__ void __launch_bounds__(128)
scan_k(const unsigned short* __restrict__ xp_u, const unsigned short* __restrict__ wws_u,
       unsigned short* __restrict__ h0_u, unsigned short* __restrict__ h1_u,
       float* __restrict__ out, unsigned* __restrict__ counter) {
  const __bf16* xp  = (const __bf16*)xp_u;
  const __bf16* wws = (const __bf16*)wws_u;
  __bf16* hbuf[2]   = {(__bf16*)h0_u, (__bf16*)h1_u};

  const int wg  = blockIdx.x;
  const int tid = threadIdx.x;
  const int w   = tid >> 5;
  const int l   = tid & 31;
  const int l16 = l & 15;
  const int h8  = (l >> 4) * 8;

  __shared__ __bf16 Wp[32 * 4 * 512];     // 65536 bf16 = 128KB : Wh slice, packed
  __shared__ __bf16 As[2][64 * 128];      // 2 x 16KB activation chunks
  uint32_t asb[2] = {(uint32_t)(uintptr_t)&As[0][0], (uint32_t)(uintptr_t)&As[1][0]};

  // load this WG's Wh fragments once (kt 32..63 of packed buffer): 128KB
  {
    const char* wsrc = (const char*)(wws + (size_t)wg * 131072 + 65536);
    uint32_t wb = (uint32_t)(uintptr_t)&Wp[0];
#pragma unroll 4
    for (int i = 0; i < 64; ++i) {
      int s = i * 128 + tid;               // 0..8191 (16B units)
      async_g2l_b128(wb + (uint32_t)s * 16, wsrc + (size_t)s * 16);
    }
    wait_async_0();
    __syncthreads();
  }

  float creg[8];
#pragma unroll
  for (int r = 0; r < 8; ++r) creg[r] = 0.0f;

  for (int t = 0; t < NT; ++t) {
    const __bf16* hin = hbuf[t & 1];
    __bf16* hout      = hbuf[(t + 1) & 1];

    // kick off first h chunk while initializing accumulators from x_proj
    stage_tile(asb[0], (const char*)hin, 2048, tid);

    v8f acc[4];
#pragma unroll
    for (int ct = 0; ct < 4; ++ct) {
      size_t idx = ((((size_t)t * 64 + wg) * 4 + w) * 4 + ct) * 256 + (size_t)l * 8;
      v8bf xv = *(const v8bf*)&xp[idx];
#pragma unroll
      for (int r = 0; r < 8; ++r) acc[ct][r] = (float)xv[r];
    }

    for (int c = 0; c < 8; ++c) {          // K = 1024 (h side) in 128-chunks
      if (c < 7) {
        stage_tile(asb[(c + 1) & 1], (const char*)hin + (size_t)(c + 1) * 256, 2048, tid);
        wait_async_le8();
      } else {
        wait_async_0();
      }
      __syncthreads();
      const __bf16* A = &As[c & 1][0];
#pragma unroll
      for (int kt = 0; kt < 4; ++kt) {
        int row = w * 16 + l16;
        V16 a;
        a.h[0] = *(const v8bf*)&A[row * 128 + kt * 32 + h8];
        a.h[1] = *(const v8bf*)&A[row * 128 + kt * 32 + h8 + 16];
        int ktl = c * 4 + kt;              // 0..31 within LDS-resident Wh
#pragma unroll
        for (int ct = 0; ct < 4; ++ct) {
          size_t bo = ((size_t)(ktl * 4 + ct)) * 512 + (size_t)l * 16;
          V16 b;
          b.h[0] = *(const v8bf*)&Wp[bo];
          b.h[1] = *(const v8bf*)&Wp[bo + 8];
          acc[ct] = wmma_bf16(a.v, b.v, acc[ct]);
        }
      }
      __syncthreads();
    }

    // pointwise gates; c lives in registers, h broadcast as bf16
#pragma unroll
    for (int r = 0; r < 8; ++r) {
      float it = sigmoid_f(acc[0][r]);
      float ft = sigmoid_f(acc[1][r]);
      float gt = tanh_f(acc[2][r]);
      float ot = sigmoid_f(acc[3][r]);
      float cn = ft * creg[r] + it * gt;
      creg[r] = cn;
      float hn = ot * tanh_f(cn);
      int m   = w * 16 + h8 + r;
      int col = wg * 16 + l16;
      hout[m * 1024 + col] = (__bf16)hn;
      if (t == NT - 1) {
        out[m * 1024 + col]         = hn;       // h_t
        out[65536 + m * 1024 + col] = creg[r];  // c_t
      }
    }

    // device-wide step barrier (monotonic counter; h double-buffered)
    __threadfence();
    __syncthreads();
    if (tid == 0) {
      __hip_atomic_fetch_add(counter, 1u, __ATOMIC_RELEASE, __HIP_MEMORY_SCOPE_AGENT);
      unsigned tgt = (unsigned)(t + 1) * NWG;
      while (__hip_atomic_load(counter, __ATOMIC_ACQUIRE, __HIP_MEMORY_SCOPE_AGENT) < tgt)
        __builtin_amdgcn_s_sleep(1);
    }
    __syncthreads();
  }
}

// ---------------------------------------------------------------------------
extern "C" void kernel_launch(void* const* d_in, const int* in_sizes, int n_in,
                              void* d_out, int out_size, void* d_ws, size_t ws_size,
                              hipStream_t stream) {
  (void)in_sizes; (void)n_in; (void)out_size; (void)ws_size;
  const float* x   = (const float*)d_in[0];
  const float* wii = (const float*)d_in[1];
  const float* wif = (const float*)d_in[2];
  const float* wig = (const float*)d_in[3];
  const float* wio = (const float*)d_in[4];
  const float* whi = (const float*)d_in[5];
  const float* whf = (const float*)d_in[6];
  const float* whg = (const float*)d_in[7];
  const float* who = (const float*)d_in[8];
  const float* bii = (const float*)d_in[9];
  const float* bif = (const float*)d_in[10];
  const float* big = (const float*)d_in[11];
  const float* bio = (const float*)d_in[12];
  const float* bhi = (const float*)d_in[13];
  const float* bhf = (const float*)d_in[14];
  const float* bhg = (const float*)d_in[15];
  const float* bho = (const float*)d_in[16];

  char* ws = (char*)d_ws;
  unsigned short* wws  = (unsigned short*)(ws + OFF_W);
  float*          bias = (float*)(ws + OFF_BIAS);
  unsigned short* xb   = (unsigned short*)(ws + OFF_XB);
  unsigned short* xp   = (unsigned short*)(ws + OFF_XP);
  unsigned short* h0   = (unsigned short*)(ws + OFF_H0);
  unsigned short* h1   = (unsigned short*)(ws + OFF_H1);
  unsigned*       cnt  = (unsigned*)(ws + OFF_CNT);
  float*          out  = (float*)d_out;

  pack_w_k<<<32768, 256, 0, stream>>>(wii, wif, wig, wio, whi, whf, whg, who, wws);
  bias_k<<<16, 256, 0, stream>>>(bii, bif, big, bio, bhi, bhf, bhg, bho, bias);
  xcvt_k<<<32768, 256, 0, stream>>>(x, xb);           // 4 elems / thread
  init_k<<<257, 256, 0, stream>>>(h0, cnt);
  xproj_k<<<16384, 128, 0, stream>>>(xb, wws, bias, xp);
  scan_k<<<NWG, 128, 0, stream>>>(xp, wws, h0, h1, out, cnt);
}
